// EpipolarCrossAttention_23235773071386
// MI455X (gfx1250) — matrix operand order repaired and checked by reading, not compiled, not run on hardware
//
#include <hip/hip_runtime.h>
#include <hip/hip_bf16.h>
#include <hip/hip_fp16.h>

typedef __attribute__((ext_vector_type(16))) _Float16 v16h;
typedef __attribute__((ext_vector_type(8)))  float    v8f;
typedef __attribute__((ext_vector_type(4)))  int      v4i;

#define BATCH 8
#define L1    2048
#define L2    2048
#define CDIM  320
#define HEADS 8
#define DHEAD 64
#define INNER 512
#define NREG  4
#define LK    2052
#define LKP   2080      /* padded to 65*32 */
#define NKW   65        /* 32-key mask words per q row */
#define SCALE 0.125f    /* 64^-0.5 */

// ---------------- async global->LDS copy (CDNA5 ASYNCcnt path) ----------------
// 16 bytes per lane, written directly to LDS without a VGPR round trip.
// Builtin signature (from hipcc diagnostic): params are raw `int __vector(4)*`.

__device__ __forceinline__ void async_cp16(const _Float16* g, _Float16* l) {
#if __has_builtin(__builtin_amdgcn_global_load_async_to_lds_b128)
  __builtin_amdgcn_global_load_async_to_lds_b128((v4i*)g, (v4i*)l, 0, 0);
#else
  *(int4*)l = *(const int4*)g;
#endif
}

__device__ __forceinline__ void wait_async() {
#if __has_builtin(__builtin_amdgcn_global_load_async_to_lds_b128) && \
    __has_builtin(__builtin_amdgcn_s_wait_asynccnt)
  __builtin_amdgcn_s_wait_asynccnt(0);
#endif
}

// ---------------- WMMA fragment helpers (ISA 7.12.2 layouts) ----------------

__device__ __forceinline__ v8f wmma16x16x32(v16h a, v16h b, v8f c) {
  // D = A(16x32 f16) * B(32x16 f16) + C(16x16 f32)
  return __builtin_amdgcn_wmma_f32_16x16x32_f16(false, a, false, b,
                                                (short)0, c, false, false);
}

// A-matrix 16x32 f16: lane row = l&15; VGPR r<4 holds k=2r+8h, r>=4 holds 16+2(r-4)+8h
__device__ __forceinline__ v16h ldfragA(const _Float16* row, int kh) {
  union { int4 i[2]; v16h v; } u;
  u.i[0] = *(const int4*)(row + 8 * kh);
  u.i[1] = *(const int4*)(row + 16 + 8 * kh);
  return u.v;
}

// B-matrix 32x16 f16 read from a [N][K] (contraction-contiguous) row:
// lane col = l&15; VGPR r<4 holds k=2r+8h, r>=4 holds 16+2(r-4)+8h (same as A)
__device__ __forceinline__ v16h ldfragB(const _Float16* row, int kh) {
  union { int4 i[2]; v16h v; } u;
  const int4* p = (const int4*)(row + 8 * kh);
  u.i[0] = p[0];
  u.i[1] = p[2];
  return u.v;
}

// ---------------- prep kernels ----------------

__global__ void k_f32_to_f16(const float* __restrict__ src,
                             _Float16* __restrict__ dst, int n) {
  int i = blockIdx.x * blockDim.x + threadIdx.x;
  if (i < n) dst[i] = (_Float16)src[i];
}

// ctx_h[b][row][c] : row<4 -> reg tokens, row<2052 -> context, else 0
__global__ void k_build_ctx(const float* __restrict__ context,
                            const float* __restrict__ reg,
                            _Float16* __restrict__ ctx) {
  int i = blockIdx.x * blockDim.x + threadIdx.x;
  if (i >= BATCH * LKP * CDIM) return;
  int c   = i % CDIM;
  int row = (i / CDIM) % LKP;
  int b   = i / (CDIM * LKP);
  float v = 0.0f;
  if (row < NREG)      v = reg[row * CDIM + c];
  else if (row < LK)   v = context[((size_t)b * L2 + (row - NREG)) * CDIM + c];
  ctx[i] = (_Float16)v;
}

// WT[n*K+k] = (f16) W[k*N+n]
__global__ void k_transpose_w(const float* __restrict__ W,
                              _Float16* __restrict__ WT, int K, int N) {
  int i = blockIdx.x * blockDim.x + threadIdx.x;
  if (i >= K * N) return;
  int k = i % K, n = i / K;
  WT[(size_t)n * K + k] = (_Float16)W[(size_t)k * N + n];
}

// bit j of word (b,q,w): key = w*32+j visible? reg tokens always, pad never
__global__ void k_pack_mask(const unsigned char* __restrict__ mask,
                            unsigned int* __restrict__ bits) {
  int i = blockIdx.x * blockDim.x + threadIdx.x;
  if (i >= BATCH * L1 * NKW) return;
  int w = i % NKW;
  int q = (i / NKW) % L1;
  int b = i / (NKW * L1);
  unsigned int v = 0;
  const unsigned char* mrow = mask + ((size_t)b * L1 + q) * L2;
#pragma unroll
  for (int j = 0; j < 32; ++j) {
    int key = w * 32 + j;
    unsigned int vis;
    if (key < NREG)    vis = 1u;
    else if (key < LK) vis = mrow[key - NREG] ? 1u : 0u;
    else               vis = 0u;
    v |= vis << j;
  }
  bits[i] = v;
}

// ---------------- WMMA GEMM: out = A[M,K] * Bt[N,K]^T ----------------
// Double-buffered async global->LDS staging over the K loop.
// MODE 0: store f16 row-major [M,N] (alpha applied)   (Q, K projections)
// MODE 1: store f16 V-transposed [B, N(=h*64+d), LKP] (V projection)
// MODE 2: store f32 [M,N] + bias                       (output projection)

template <int MODE>
__global__ __launch_bounds__(256, 1)
void k_gemm(const _Float16* __restrict__ A, const _Float16* __restrict__ Bt,
            void* __restrict__ out, int M, int N, int K, float alpha,
            const float* __restrict__ bias) {
  __shared__ _Float16 At[2][64][32];
  __shared__ _Float16 Bts[2][128][32];

  const int tid   = threadIdx.x;
  const int mbase = blockIdx.x * 64;
  const int nbase = blockIdx.y * 128;
  const int wave  = tid >> 5;
  const int lane  = tid & 31;
  const int ln    = lane & 15;
  const int kh    = lane >> 4;
  const int wm    = wave & 3;   // 4 M sub-tiles of 16
  const int wn    = wave >> 2;  // 2 N sub-tiles of 64

  v8f acc[4];
#pragma unroll
  for (int t = 0; t < 4; ++t)
#pragma unroll
    for (int r = 0; r < 8; ++r) acc[t][r] = 0.0f;

  const int arow = tid >> 2, ach = (tid & 3) * 8;      // A stage: 1x 16B / thread
  const int brow = tid >> 1, bch = (tid & 1) * 16;     // Bt stage: 2x 16B / thread
  const int bn   = nbase + brow;

  auto stage = [&](int kb, int buf) {
    async_cp16(A + (size_t)(mbase + arow) * K + kb + ach, &At[buf][arow][ach]);
    if (bn < N) {
      const _Float16* src = Bt + (size_t)bn * K + kb + bch;
      async_cp16(src,     &Bts[buf][brow][bch]);
      async_cp16(src + 8, &Bts[buf][brow][bch + 8]);
    } else {
      int4 z = {0, 0, 0, 0};
      *(int4*)&Bts[buf][brow][bch]     = z;
      *(int4*)&Bts[buf][brow][bch + 8] = z;
    }
  };

  stage(0, 0);
  const int niter = K / 32;
  for (int it = 0; it < niter; ++it) {
    const int cur = it & 1;
    wait_async();        // my async copies into buf `cur` have landed
    __syncthreads();     // everyone's have; previous readers of `cur^1` done
    if (it + 1 < niter) stage((it + 1) * 32, cur ^ 1);  // prefetch next tile

    v16h a = ldfragA(&At[cur][wm * 16 + ln][0], kh);
#pragma unroll
    for (int t = 0; t < 4; ++t) {
      v16h b = ldfragB(&Bts[cur][wn * 64 + t * 16 + ln][0], kh);
      acc[t] = wmma16x16x32(a, b, acc[t]);
    }
    __syncthreads();     // done reading `cur` before it is restaged
  }

#pragma unroll
  for (int t = 0; t < 4; ++t) {
#pragma unroll
    for (int r = 0; r < 8; ++r) {
      int m = mbase + wm * 16 + r + 8 * kh;
      int n = nbase + wn * 64 + t * 16 + ln;
      if (n >= N) continue;
      float v = acc[t][r] * alpha;
      if (MODE == 0) {
        ((_Float16*)out)[(size_t)m * N + n] = (_Float16)v;
      } else if (MODE == 1) {
        int b  = m / LKP;
        int kv = m % LKP;
        ((_Float16*)out)[((size_t)b * INNER + n) * LKP + kv] = (_Float16)v;
      } else {
        ((float*)out)[(size_t)m * N + n] = v + bias[n];
      }
    }
  }
}

// ---------------- flash attention ----------------
// grid (32 q-tiles, 64 b*h), 128 threads (4 waves x 16 q rows).
// Double-buffered async staging of K/V tiles: DMA tile kt+1 while computing kt.

__global__ __launch_bounds__(128, 1)
void k_attn(const _Float16* __restrict__ Qh, const _Float16* __restrict__ Kh,
            const _Float16* __restrict__ Vt,
            const unsigned int* __restrict__ mbits, _Float16* __restrict__ Oh) {
  __shared__ _Float16 kt_lds[2][32][64];   // [key][d]  = B-op for Q*K^T
  __shared__ _Float16 vt_lds[2][64][32];   // [d][key]  = B-op for P*V
  __shared__ _Float16 p_lds[4][16][32];    // per-wave P tile (C->A relayout)

  const int qt   = blockIdx.x;
  const int bh   = blockIdx.y;
  const int b    = bh >> 3, h = bh & 7;
  const int tid  = threadIdx.x;
  const int wave = tid >> 5;
  const int lane = tid & 31;
  const int ln   = lane & 15;
  const int kh   = lane >> 4;
  const int qbase = qt * 64 + wave * 16;

  // Q fragments resident in VGPRs (softmax scale folded in at projection)
  const _Float16* qrow = Qh + ((size_t)(b * L1 + qbase + ln)) * INNER + h * DHEAD;
  const v16h aQ0 = ldfragA(qrow, kh);       // d 0..31
  const v16h aQ1 = ldfragA(qrow + 32, kh);  // d 32..63

  v8f o[4];
  float mrow[8], lrow[8];
#pragma unroll
  for (int t = 0; t < 4; ++t)
#pragma unroll
    for (int r = 0; r < 8; ++r) o[t][r] = 0.0f;
#pragma unroll
  for (int r = 0; r < 8; ++r) { mrow[r] = -1e30f; lrow[r] = 0.0f; }

  const unsigned int* mb = mbits + ((size_t)(b * L1 + qbase + 8 * kh)) * NKW;

  // staging maps (128 threads, 2x 16B each per tile)
  const int krow = tid >> 2, kch = (tid & 3) * 16;  // K tile 32x64
  const int vrow = tid >> 1, vch = (tid & 1) * 16;  // Vt tile 64x32

  auto stage = [&](int kt, int buf) {
    const int kb = kt * 32;
    const _Float16* ks =
        Kh + ((size_t)(b * LKP + kb + krow)) * INNER + h * DHEAD + kch;
    async_cp16(ks,     &kt_lds[buf][krow][kch]);
    async_cp16(ks + 8, &kt_lds[buf][krow][kch + 8]);
    const _Float16* vs =
        Vt + ((size_t)(b * INNER + h * DHEAD + vrow)) * LKP + kb + vch;
    async_cp16(vs,     &vt_lds[buf][vrow][vch]);
    async_cp16(vs + 8, &vt_lds[buf][vrow][vch + 8]);
  };

  stage(0, 0);
  for (int kt = 0; kt < NKW; ++kt) {
    const int cur = kt & 1;
    wait_async();        // my copies into buf `cur` landed
    __syncthreads();     // all waves' copies landed; prior readers of it done
    if (kt + 1 < NKW) stage(kt + 1, cur ^ 1);   // DMA next tile during compute

    unsigned int mw[8];
#pragma unroll
    for (int r = 0; r < 8; ++r) mw[r] = mb[r * NKW + kt];

    // scores: two 16x16 key tiles, contraction over d=64 (2 WMMAs each)
    v8f s0, s1;
#pragma unroll
    for (int r = 0; r < 8; ++r) { s0[r] = 0.0f; s1[r] = 0.0f; }
    {
      v16h b00 = ldfragB(&kt_lds[cur][ln][0], kh);
      v16h b01 = ldfragB(&kt_lds[cur][ln][32], kh);
      s0 = wmma16x16x32(aQ0, b00, s0);
      s0 = wmma16x16x32(aQ1, b01, s0);
      v16h b10 = ldfragB(&kt_lds[cur][16 + ln][0], kh);
      v16h b11 = ldfragB(&kt_lds[cur][16 + ln][32], kh);
      s1 = wmma16x16x32(aQ0, b10, s1);
      s1 = wmma16x16x32(aQ1, b11, s1);
    }

    // mask + online softmax; row r+8*kh lives in this lane's 16-lane group
#pragma unroll
    for (int r = 0; r < 8; ++r) {
      float x0 = ((mw[r] >> ln) & 1u) ? s0[r] : -1e30f;
      float x1 = ((mw[r] >> (16 + ln)) & 1u) ? s1[r] : -1e30f;
      float mx = fmaxf(x0, x1);
#pragma unroll
      for (int d = 1; d < 16; d <<= 1) mx = fmaxf(mx, __shfl_xor(mx, d, 32));
      float mnew = fmaxf(mrow[r], mx);
      float corr = __expf(mrow[r] - mnew);
      float e0 = __expf(x0 - mnew);
      float e1 = __expf(x1 - mnew);
      float rs = e0 + e1;
#pragma unroll
      for (int d = 1; d < 16; d <<= 1) rs += __shfl_xor(rs, d, 32);
      lrow[r] = lrow[r] * corr + rs;
      mrow[r] = mnew;
#pragma unroll
      for (int t = 0; t < 4; ++t) o[t][r] *= corr;
      p_lds[wave][r + 8 * kh][ln]      = (_Float16)e0;
      p_lds[wave][r + 8 * kh][16 + ln] = (_Float16)e1;
    }
    __builtin_amdgcn_wave_barrier();  // same-wave LDS is in-order (DScnt)

    // O += P(16x32) * V(32x64): 4 WMMAs
    {
      v16h aP = ldfragA(&p_lds[wave][ln][0], kh);
#pragma unroll
      for (int t = 0; t < 4; ++t) {
        v16h bv = ldfragB(&vt_lds[cur][t * 16 + ln][0], kh);
        o[t] = wmma16x16x32(aP, bv, o[t]);
      }
    }
    __syncthreads();     // done reading `cur` before it is restaged
  }

  // normalize + store f16 O [q][h*64+d]
#pragma unroll
  for (int r = 0; r < 8; ++r) {
    float inv = 1.0f / lrow[r];
    int q = qbase + r + 8 * kh;
    _Float16* orow = Oh + ((size_t)(b * L1 + q)) * INNER + h * DHEAD;
#pragma unroll
    for (int t = 0; t < 4; ++t) orow[t * 16 + ln] = (_Float16)(o[t][r] * inv);
  }
}

// ---------------- host-side launch ----------------

static inline size_t align256(size_t x) { return (x + 255) & ~(size_t)255; }

extern "C" void kernel_launch(void* const* d_in, const int* in_sizes, int n_in,
                              void* d_out, int out_size, void* d_ws, size_t ws_size,
                              hipStream_t stream) {
  const float*         x    = (const float*)d_in[0];
  const float*         ctx  = (const float*)d_in[1];
  const unsigned char* mask = (const unsigned char*)d_in[2];
  const float*         Wq   = (const float*)d_in[3];
  const float*         Wk   = (const float*)d_in[4];
  const float*         Wv   = (const float*)d_in[5];
  const float*         Wo   = (const float*)d_in[6];
  const float*         bo   = (const float*)d_in[7];
  const float*         reg  = (const float*)d_in[8];

  char* ws = (char*)d_ws;
  size_t off = 0;
  auto bump = [&](size_t bytes) { size_t o = off; off = align256(off + bytes); return o; };

  _Float16* Xh   = (_Float16*)(ws + bump((size_t)BATCH * L1 * CDIM * 2));
  _Float16* CTXh = (_Float16*)(ws + bump((size_t)BATCH * LKP * CDIM * 2));
  _Float16* WqT  = (_Float16*)(ws + bump((size_t)INNER * CDIM * 2));
  _Float16* WkT  = (_Float16*)(ws + bump((size_t)INNER * CDIM * 2));
  _Float16* WvT  = (_Float16*)(ws + bump((size_t)INNER * CDIM * 2));
  _Float16* WoT  = (_Float16*)(ws + bump((size_t)CDIM * INNER * 2));
  _Float16* Qh   = (_Float16*)(ws + bump((size_t)BATCH * L1 * INNER * 2));
  _Float16* Kh   = (_Float16*)(ws + bump((size_t)BATCH * LKP * INNER * 2));
  _Float16* Vt   = (_Float16*)(ws + bump((size_t)BATCH * INNER * LKP * 2));
  _Float16* Oh   = (_Float16*)(ws + bump((size_t)BATCH * L1 * INNER * 2));
  unsigned int* mbits = (unsigned int*)(ws + bump((size_t)BATCH * L1 * NKW * 4));
  (void)ws_size; (void)in_sizes; (void)n_in; (void)out_size;

  // ---- prep ----
  {
    int n = BATCH * L1 * CDIM;
    k_f32_to_f16<<<(n + 255) / 256, 256, 0, stream>>>(x, Xh, n);
  }
  {
    int n = BATCH * LKP * CDIM;
    k_build_ctx<<<(n + 255) / 256, 256, 0, stream>>>(ctx, reg, CTXh);
  }
  {
    int n = CDIM * INNER;
    k_transpose_w<<<(n + 255) / 256, 256, 0, stream>>>(Wq, WqT, CDIM, INNER);
    k_transpose_w<<<(n + 255) / 256, 256, 0, stream>>>(Wk, WkT, CDIM, INNER);
    k_transpose_w<<<(n + 255) / 256, 256, 0, stream>>>(Wv, WvT, CDIM, INNER);
    k_transpose_w<<<(n + 255) / 256, 256, 0, stream>>>(Wo, WoT, INNER, CDIM);
  }
  {
    int n = BATCH * L1 * NKW;
    k_pack_mask<<<(n + 255) / 256, 256, 0, stream>>>(mask, mbits);
  }

  // ---- projections (WMMA GEMMs) ----
  k_gemm<0><<<dim3((BATCH * L1) / 64, INNER / 128), 256, 0, stream>>>(
      Xh, WqT, Qh, BATCH * L1, INNER, CDIM, SCALE, nullptr);
  k_gemm<0><<<dim3((BATCH * LKP) / 64, INNER / 128), 256, 0, stream>>>(
      CTXh, WkT, Kh, BATCH * LKP, INNER, CDIM, 1.0f, nullptr);
  k_gemm<1><<<dim3((BATCH * LKP) / 64, INNER / 128), 256, 0, stream>>>(
      CTXh, WvT, Vt, BATCH * LKP, INNER, CDIM, 1.0f, nullptr);

  // ---- flash attention ----
  k_attn<<<dim3(L1 / 64, BATCH * HEADS), 128, 0, stream>>>(Qh, Kh, Vt, mbits, Oh);

  // ---- output projection + bias (f32 out) ----
  k_gemm<2><<<dim3((BATCH * L1) / 64, (CDIM + 127) / 128), 256, 0, stream>>>(
      Oh, WoT, d_out, BATCH * L1, CDIM, INNER, 1.0f, bo);
}
